// SLLinear_47304769798216
// MI455X (gfx1250) — matrix operand order repaired and checked by reading, not compiled, not run on hardware
//
#include <hip/hip_runtime.h>
#include <hip/hip_bf16.h>
#include <math.h>

// ---------------------------------------------------------------------------
// SLLinear on gfx1250 (MI455X): bf16 WMMA, fused gate + t=x@L1 + masked GEMM
// + low-rank GEMM + sigmoid-gated epilogue. Wave32 layouts per CDNA5 ISA 7.12.2.
// ---------------------------------------------------------------------------

typedef __attribute__((ext_vector_type(16))) __bf16 v16bf;
typedef __attribute__((ext_vector_type(8)))  __bf16 v8bf;
typedef __attribute__((ext_vector_type(8)))  float  v8f;

#define IN_F   1024
#define OUT_F  1024
#define RANK   102
#define RANKP  128                 // rank padded to multiple of 32
#define MROWS  32768               // 8 * 4096
#define MT     (MROWS / 16)        // 2048 M-tiles
#define KT     (IN_F / 32)         // 32 k-steps
#define KT_LR  (RANKP / 32)        // 4 k-steps for low-rank GEMM
#define TLS_STRIDE 136             // 128 + 8 pad: conflict-free b128 LDS reads

union A16u { v16bf v; struct { v8bf lo, hi; } p; };

static __device__ __forceinline__ v8f wmma_bf16(v16bf a, v16bf b, v8f c) {
  return __builtin_amdgcn_wmma_f32_16x16x32_bf16(
      false, a, false, b, (short)0, c, false, false);
}

// --- zero the gate accumulator --------------------------------------------
__global__ void k_zero_g(float* __restrict__ gacc) {
  unsigned i = blockIdx.x * blockDim.x + threadIdx.x;
  if (i < MROWS) gacc[i] = 0.0f;
}

// --- pack x -> bf16 A layout, fused gate partial dot ----------------------
// tile (mt,kt): lane l holds row mt*16+(l&15), k=kb..kb+15
// with kb = kt*32 + (l>>4)*16.  Gate partials reduced via shfl + atomicAdd.
__global__ void k_pack_x(const float* __restrict__ x, const float* __restrict__ gw,
                         __bf16* __restrict__ xpk, float* __restrict__ gacc) {
  unsigned tid = blockIdx.x * blockDim.x + threadIdx.x;   // mt*1024 + kt*32 + lane
  unsigned lane = tid & 31u, kt = (tid >> 5) & 31u, mt = tid >> 10;
  unsigned row = mt * 16 + (lane & 15u);
  unsigned kb  = kt * 32 + ((lane >> 4) << 4);
  const float* src = x + (size_t)row * IN_F + kb;
  float p = 0.0f;
  v16bf a;
#pragma unroll
  for (int h = 0; h < 8; ++h) {
    float v0 = src[h];
    a[h] = (__bf16)v0;
    p = fmaf(v0, gw[kb + h], p);
  }
#pragma unroll
  for (int h = 0; h < 8; ++h) {
    float v1 = src[8 + h];
    a[8 + h] = (__bf16)v1;
    p = fmaf(v1, gw[kb + 8 + h], p);
  }
  *(v16bf*)(xpk + (size_t)tid * 16) = a;
  p += __shfl_xor(p, 16, 32);          // fold the two half-wave k-slices
  if (lane < 16) atomicAdd(&gacc[row], p);
}

// --- finish gate: sigmoid --------------------------------------------------
__global__ void k_gate_fin(const float* __restrict__ gacc, const float* __restrict__ gb,
                           float* __restrict__ g) {
  unsigned i = blockIdx.x * blockDim.x + threadIdx.x;
  if (i < MROWS) g[i] = 1.0f / (1.0f + expf(-(gacc[i] + gb[0])));
}

// --- pack mask*weight -> bf16 B layout -------------------------------------
__global__ void k_pack_w(const float* __restrict__ w, const float* __restrict__ m,
                         __bf16* __restrict__ wpk) {
  unsigned tid = blockIdx.x * blockDim.x + threadIdx.x;   // nt*1024 + kt*32 + lane
  unsigned lane = tid & 31u, kt = (tid >> 5) & 31u, nt = tid >> 10;
  unsigned n  = nt * 16 + (lane & 15u);
  unsigned kb = kt * 32 + ((lane >> 4) << 4);
  const float* wr = w + (size_t)n * IN_F + kb;
  const float* mr = m + (size_t)n * IN_F + kb;
  v16bf b;
#pragma unroll
  for (int j = 0; j < 16; ++j) b[j] = (__bf16)(wr[j] * mr[j]);
  *(v16bf*)(wpk + (size_t)tid * 16) = b;
}

// --- pack low_rank1 [1024 x 102] -> B layout, N padded to 128 --------------
__global__ void k_pack_l1(const float* __restrict__ l1, __bf16* __restrict__ l1pk) {
  unsigned tid = blockIdx.x * blockDim.x + threadIdx.x;   // nt*1024 + kt*32 + lane
  unsigned lane = tid & 31u, kt = (tid >> 5) & 31u, nt = tid >> 10;  // nt 0..7
  unsigned n  = nt * 16 + (lane & 15u);
  unsigned kb = kt * 32 + ((lane >> 4) << 4);
  v16bf b;
#pragma unroll
  for (int j = 0; j < 16; ++j)
    b[j] = (n < RANK) ? (__bf16)l1[(size_t)(kb + j) * RANK + n] : (__bf16)0.0f;
  *(v16bf*)(l1pk + (size_t)tid * 16) = b;
}

// --- pack low_rank2 [102 x 1024] -> B layout, K padded to 128 --------------
__global__ void k_pack_l2(const float* __restrict__ l2, __bf16* __restrict__ l2pk) {
  unsigned tid = blockIdx.x * blockDim.x + threadIdx.x;   // nt*128 + kt*32 + lane
  unsigned lane = tid & 31u, kt = (tid >> 5) & 3u, nt = tid >> 7;    // kt 0..3, nt 0..63
  unsigned n  = nt * 16 + (lane & 15u);
  unsigned kb = kt * 32 + ((lane >> 4) << 4);
  v16bf b;
#pragma unroll
  for (int j = 0; j < 16; ++j) {
    unsigned k = kb + j;
    b[j] = (k < RANK) ? (__bf16)l2[(size_t)k * OUT_F + n] : (__bf16)0.0f;
  }
  *(v16bf*)(l2pk + (size_t)tid * 16) = b;
}

// --- main: one M-tile per block, full N per block --------------------------
// wave w handles N-tiles w*8..w*8+7 (accS[8]) and also computes rank-tile w of
// t = x@L1 in the same k-loop (9 WMMAs/step).  t goes through LDS (transpose
// D-layout -> A-layout), then the low-rank GEMM + gated epilogue per N-tile.
__global__ void __launch_bounds__(256) k_main(
    const __bf16* __restrict__ xpk, const __bf16* __restrict__ wpk,
    const __bf16* __restrict__ l1pk, const __bf16* __restrict__ l2pk,
    const float* __restrict__ g, const float* __restrict__ bs,
    const float* __restrict__ blr, float* __restrict__ out) {
  __shared__ __align__(32) __bf16 tls[16 * TLS_STRIDE];

  unsigned lane = threadIdx.x & 31u;
  unsigned w    = threadIdx.x >> 5;      // wave 0..7
  unsigned mt   = blockIdx.x;            // 0..2047

  const v16bf* av  = (const v16bf*)xpk  + (size_t)mt * (KT * 32) + lane;
  const v16bf* bv  = (const v16bf*)wpk  + lane;
  const v16bf* l1v = (const v16bf*)l1pk + (size_t)w * (KT * 32) + lane;

  v8f accS[8] = {};
  v8f accT    = {};

  // fused masked-GEMM + t-GEMM k-loop: A loaded once, 9 WMMAs reuse it
#pragma unroll 2
  for (int kt = 0; kt < KT; ++kt) {
    v16bf a = av[kt * 32];
    accT = wmma_bf16(a, l1v[kt * 32], accT);
#pragma unroll
    for (int j = 0; j < 8; ++j) {
      v16bf b = bv[((size_t)(w * 8 + j) * KT + kt) * 32];
      accS[j] = wmma_bf16(a, b, accS[j]);
    }
  }

  // stash t tile (rank columns w*16..w*16+15) into LDS as bf16
  {
    unsigned rc = w * 16 + (lane & 15u);         // rank column
    unsigned r8 = (lane >> 4) << 3;              // row base within M-tile
#pragma unroll
    for (int v = 0; v < 8; ++v)
      tls[(r8 + v) * TLS_STRIDE + rc] = (__bf16)accT[v];
  }
  __syncthreads();

  // re-read t in WMMA A-operand layout (16B-aligned ds_load_b128 pairs)
  A16u aL[KT_LR];
  {
    const __bf16* trow = tls + (size_t)(lane & 15u) * TLS_STRIDE;
    unsigned kb = (lane >> 4) << 4;
#pragma unroll
    for (int kt = 0; kt < KT_LR; ++kt) {
      aL[kt].p.lo = *(const v8bf*)(trow + kt * 32 + kb);
      aL[kt].p.hi = *(const v8bf*)(trow + kt * 32 + kb + 8);
    }
  }

  unsigned rbase = mt * 16 + ((lane >> 4) << 3);
  float gv[8];
#pragma unroll
  for (int v = 0; v < 8; ++v) gv[v] = g[rbase + v];

  const v16bf* l2v = (const v16bf*)l2pk + lane;
#pragma unroll
  for (int j = 0; j < 8; ++j) {
    v8f accL = {};
#pragma unroll
    for (int kt = 0; kt < KT_LR; ++kt)
      accL = wmma_bf16(aL[kt].v, l2v[((size_t)(w * 8 + j) * KT_LR + kt) * 32], accL);
    unsigned n = (w * 8 + j) * 16 + (lane & 15u);
    float b0 = bs[n], b1 = blr[n];
#pragma unroll
    for (int v = 0; v < 8; ++v) {
      float gg = gv[v];
      out[(size_t)(rbase + v) * OUT_F + n] =
          gg * (accS[j][v] + b0) + (1.0f - gg) * (accL[v] + b1);
    }
  }
}

// ---------------------------------------------------------------------------
extern "C" void kernel_launch(void* const* d_in, const int* in_sizes, int n_in,
                              void* d_out, int out_size, void* d_ws, size_t ws_size,
                              hipStream_t stream) {
  const float* x    = (const float*)d_in[0];
  const float* gw   = (const float*)d_in[1];
  const float* gb   = (const float*)d_in[2];
  const float* wgt  = (const float*)d_in[3];
  const float* bsp  = (const float*)d_in[4];
  const float* l1   = (const float*)d_in[5];
  const float* l2   = (const float*)d_in[6];
  const float* blr  = (const float*)d_in[7];
  const float* msk  = (const float*)d_in[8];
  float* out = (float*)d_out;

  // workspace carve-up (~66.8 MiB)
  char* ws = (char*)d_ws;
  __bf16* xpk  = (__bf16*)(ws);                         // 64 MiB
  __bf16* wpk  = (__bf16*)(ws + (size_t)67108864);      //  2 MiB
  __bf16* l1pk = (__bf16*)(ws + (size_t)69206016);      // 256 KiB
  __bf16* l2pk = (__bf16*)(ws + (size_t)69468160);      // 256 KiB
  float*  gacc = (float*) (ws + (size_t)69730304);      // 128 KiB
  float*  gbuf = (float*) (ws + (size_t)69861376);      // 128 KiB

  k_zero_g <<<128,  256, 0, stream>>>(gacc);
  k_pack_x <<<8192, 256, 0, stream>>>(x, gw, xpk, gacc);
  k_pack_w <<<256,  256, 0, stream>>>(wgt, msk, wpk);
  k_pack_l1<<<32,   256, 0, stream>>>(l1, l1pk);
  k_pack_l2<<<32,   256, 0, stream>>>(l2, l2pk);
  k_gate_fin<<<128, 256, 0, stream>>>(gacc, gb, gbuf);

  k_main<<<MT, 256, 0, stream>>>(xpk, wpk, l1pk, l2pk, gbuf, bsp, blr, out);
}
